// CrfDecoder_4964982194325
// MI455X (gfx1250) — matrix operand order repaired and checked by reading, not compiled, not run on hardware
//
#include <hip/hip_runtime.h>

typedef __attribute__((ext_vector_type(16))) _Float16 v16h;
typedef __attribute__((ext_vector_type(8)))  _Float16 v8h;
typedef __attribute__((ext_vector_type(8)))  float    v8f;

#define BB   64
#define TT   2048
#define NN   128
#define ROWS 16          // batch rows per workgroup (WMMA M)
#define PITCH 136        // padded row pitch (elements) for LDS arrays

#define LOG2E 1.44269504088896340736f
#define LN2   0.69314718055994530942f

// ---------------------------------------------------------------------------
// Prep: expT[i*N + j] = (f16) exp(transitions[i][j])   (i = prev tag, j = next)
// ---------------------------------------------------------------------------
__global__ void crf_prep_expT(const float* __restrict__ trans,
                              _Float16* __restrict__ expT) {
    int i = blockIdx.x * blockDim.x + threadIdx.x;
    if (i < NN * NN)
        expT[i] = (_Float16)__builtin_amdgcn_exp2f(trans[i] * LOG2E);
}

// ---------------------------------------------------------------------------
// Main forward scan. One workgroup = 16 batch rows, 8 wave32s; each wave owns
// one 16-wide column tile of N=128 with its expT B-fragments held in VGPRs.
// Per step (2 barriers):
//   phase 1: row-max via shfl_xor butterfly, p = exp2((alpha-m)*log2e) -> LDS
//   phase 2: S = P @ expT via 4x v_wmma_f32_16x16x32_f16,
//            alpha' = m + ln2*log2(S) + em[t]
// Time loop split at tmin = min(len): masking code only in the tail region.
// ---------------------------------------------------------------------------
__global__ __launch_bounds__(256, 1)
void crf_forward(const float* __restrict__ em,        // [B,T,1,N]
                 const int*   __restrict__ lens,      // [B]
                 const _Float16* __restrict__ expT,   // [N,N]
                 const float* __restrict__ head,      // [N]
                 const float* __restrict__ last,      // [N]
                 float* __restrict__ out)             // [B]
{
    __shared__ float    alpha[ROWS][PITCH];
    __shared__ _Float16 pbuf[ROWS][PITCH];
    __shared__ float    pm[ROWS][16];
    __shared__ float    mrow[ROWS];
    __shared__ int      len_s[ROWS];

    const int tid  = threadIdx.x;
    const int lane = tid & 31;
    const int wave = tid >> 5;            // 0..7 -> 16-wide column tile
    const int n0   = wave * 16;
    const int b0   = blockIdx.x * ROWS;

    const int prow = tid >> 4;            // 0..15  (reduction-phase row)
    const int pseg = tid & 15;            // 0..15  (8 columns each)
    const int c0   = pseg * 8;

    if (tid < ROWS) len_s[tid] = lens[b0 + tid];

    // alpha_0 = head + emissions[:,0]
    {
        const unsigned e0 = (unsigned)(b0 + prow) * (TT * NN) + c0;
        #pragma unroll
        for (int i = 0; i < 8; ++i)
            alpha[prow][c0 + i] = head[c0 + i] + em[e0 + i];
    }
    __syncthreads();

    // Preload this wave's expT B-fragments (constant across all timesteps).
    // 16-bit B layout: lane -> N = n0 + (lane&15); element e -> K = kbase + e
    // (e<8) or 16 + kbase + e-8 (e>=8), kbase = (lane<16 ? 0 : 8), per 32-deep K chunk.
    v16h bfrag[4];
    {
        const int n     = n0 + (lane & 15);
        const int kbase = (lane < 16) ? 0 : 8;
        #pragma unroll
        for (int c = 0; c < 4; ++c)
            #pragma unroll
            for (int e = 0; e < 16; ++e)
                bfrag[c][e] = expT[(32 * c + kbase + (e < 8 ? e : e + 8)) * NN + n];
    }

    // C/D layout: VGPR r -> row rbase + r, column colC (per lane)
    const int rbase = (lane < 16) ? 0 : 8;
    const int colC  = n0 + (lane & 15);

    int   len_r[8];
    float aval[8];                      // register copy of alpha[rbase+r][colC]
    #pragma unroll
    for (int r = 0; r < 8; ++r) {
        len_r[r] = len_s[rbase + r];
        aval[r]  = alpha[rbase + r][colC];
    }
    // single 32-bit running element offset; rows are r*TT*NN apart (fits the
    // signed 24-bit instruction offset: r * 1MB <= 7MB)
    unsigned ebase = ((unsigned)(b0 + rbase) * TT + 1u) * NN + colC;

    // Unmasked region bound: masking can't trigger before the block-min length.
    int tmin = TT;
    #pragma unroll
    for (int i = 0; i < ROWS; ++i) tmin = min(tmin, len_s[i]);

    auto step = [&](int t, bool masked) __attribute__((always_inline)) {
        // ---- phase 1: row max (in-wave butterfly) + p = exp(alpha - m) ----
        {
            float v[8];
            const float4* ap = (const float4*)&alpha[prow][c0];
            float4 a0 = ap[0], a1 = ap[1];
            v[0] = a0.x; v[1] = a0.y; v[2] = a0.z; v[3] = a0.w;
            v[4] = a1.x; v[5] = a1.y; v[6] = a1.z; v[7] = a1.w;
            float mx = v[0];
            #pragma unroll
            for (int i = 1; i < 8; ++i) mx = fmaxf(mx, v[i]);
            // 16 threads of a row are contiguous in one wave: butterfly reduce
            mx = fmaxf(mx, __shfl_xor(mx, 1));
            mx = fmaxf(mx, __shfl_xor(mx, 2));
            mx = fmaxf(mx, __shfl_xor(mx, 4));
            mx = fmaxf(mx, __shfl_xor(mx, 8));
            if (pseg == 0) mrow[prow] = mx;
            v8h ph;
            #pragma unroll
            for (int i = 0; i < 8; ++i)
                ph[i] = (_Float16)__builtin_amdgcn_exp2f((v[i] - mx) * LOG2E);
            *(v8h*)&pbuf[prow][c0] = ph;
        }
        __syncthreads();

        // ---- phase 2: emissions (global, overlap WMMA), WMMA, update ------
        float emv[8];
        #pragma unroll
        for (int r = 0; r < 8; ++r)
            emv[r] = em[ebase + (unsigned)(r * (TT * NN))];

        float mr[8];
        {
            const float4* mv = (const float4*)&mrow[rbase];
            float4 m0 = mv[0], m1 = mv[1];
            mr[0] = m0.x; mr[1] = m0.y; mr[2] = m0.z; mr[3] = m0.w;
            mr[4] = m1.x; mr[5] = m1.y; mr[6] = m1.z; mr[7] = m1.w;
        }

        v8f acc = {};
        {
            // 16-bit A layout: elements 0-7 = K koff..+7, 8-15 = +16,
            // koff = (lane<16 ? 0 : 8), row = lane&15 -> 2 aligned b128 reads
            const int row  = lane & 15;
            const int koff = (lane < 16) ? 0 : 8;
            #pragma unroll
            for (int c = 0; c < 4; ++c) {
                const _Float16* base = &pbuf[row][32 * c + koff];
                v16h a;
                *(v8h*)&a         = *(const v8h*)(base);
                *(((v8h*)&a) + 1) = *(const v8h*)(base + 16);
                acc = __builtin_amdgcn_wmma_f32_16x16x32_f16(
                        false, a, false, bfrag[c], (short)0, acc, false, false);
            }
        }

        // independent work first: fills the WMMA->VALU hazard window
        ebase += NN;
        float base[8];
        #pragma unroll
        for (int r = 0; r < 8; ++r) base[r] = mr[r] + emv[r];

        // acc >= exp(-|T|max) > 0.2: raw v_log_f32 safe (no denorm fixup)
        float newv[8];
        #pragma unroll
        for (int r = 0; r < 8; ++r)
            newv[r] = fmaf(LN2, __builtin_amdgcn_logf(acc[r]), base[r]);

        if (masked) {
            #pragma unroll
            for (int r = 0; r < 8; ++r) {             // pure data-flow select
                unsigned msk = (unsigned)(-(int)(t < len_r[r]));
                aval[r] = __uint_as_float((__float_as_uint(newv[r]) &  msk) |
                                          (__float_as_uint(aval[r]) & ~msk));
            }
        } else {
            #pragma unroll
            for (int r = 0; r < 8; ++r) aval[r] = newv[r];
        }
        #pragma unroll
        for (int r = 0; r < 8; ++r) alpha[rbase + r][colC] = aval[r];

        __builtin_prefetch(&em[ebase], 0, 3);         // next step's line
        __syncthreads();
    };

    for (int t = 1; t < tmin; ++t) step(t, false);    // no masking possible
    for (int t = tmin; t < TT; ++t) step(t, true);    // masked tail

    // ---- finalize: out[b] = logsumexp_j(alpha[b][j] + last[j]) ------------
    {
        float mx = -3.402823466e38f;
        #pragma unroll
        for (int i = 0; i < 8; ++i)
            mx = fmaxf(mx, alpha[prow][c0 + i] + last[c0 + i]);
        mx = fmaxf(mx, __shfl_xor(mx, 1));
        mx = fmaxf(mx, __shfl_xor(mx, 2));
        mx = fmaxf(mx, __shfl_xor(mx, 4));
        mx = fmaxf(mx, __shfl_xor(mx, 8));
        float s = 0.0f;
        #pragma unroll
        for (int i = 0; i < 8; ++i)
            s += __builtin_amdgcn_exp2f(
                    (alpha[prow][c0 + i] + last[c0 + i] - mx) * LOG2E);
        pm[prow][pseg] = s;
        if (pseg == 0) mrow[prow] = mx;
    }
    __syncthreads();
    if (tid < ROWS) {
        float s = 0.0f;
        #pragma unroll
        for (int i = 0; i < 16; ++i) s += pm[tid][i];
        out[b0 + tid] = mrow[tid] + LN2 * __builtin_amdgcn_logf(s);
    }
}

// ---------------------------------------------------------------------------
extern "C" void kernel_launch(void* const* d_in, const int* in_sizes, int n_in,
                              void* d_out, int out_size, void* d_ws, size_t ws_size,
                              hipStream_t stream) {
    const float* emissions   = (const float*)d_in[0];   // [64,2048,1,128]
    const int*   token_sizes = (const int*)  d_in[1];   // [64]
    const float* transitions = (const float*)d_in[2];   // [1,1,128,128]
    const float* head_trans  = (const float*)d_in[3];   // [1,1,128]
    const float* last_trans  = (const float*)d_in[4];   // [1,1,128]
    float*       out         = (float*)d_out;           // [64,1]

    _Float16* expT = (_Float16*)d_ws;                   // 128*128*2 = 32 KB

    crf_prep_expT<<<(NN * NN + 255) / 256, 256, 0, stream>>>(transitions, expT);
    crf_forward<<<BB / ROWS, 256, 0, stream>>>(emissions, token_sizes, expT,
                                               head_trans, last_trans, out);
}
